// CrossAttention_58033598103557
// MI455X (gfx1250) — matrix operand order repaired and checked by reading, not compiled, not run on hardware
//
#include <hip/hip_runtime.h>
#include <hip/hip_bf16.h>

#define B_    4
#define LQ_   4096
#define LKV_  4096
#define DIN_  256
#define DOUT_ 128

typedef __bf16 bf16;
typedef __attribute__((ext_vector_type(16))) __bf16 v16bf;
typedef __attribute__((ext_vector_type(8)))  __bf16 v8bf;
typedef __attribute__((ext_vector_type(8)))  float  v8f;
typedef __attribute__((ext_vector_type(4)))  unsigned int u32x4;
typedef __attribute__((ext_vector_type(8)))  int i32x8;
typedef __attribute__((ext_vector_type(4)))  int i32x4;

__device__ __forceinline__ v16bf cat8(v8bf lo, v8bf hi) {
  return __builtin_shufflevector(lo, hi, 0,1,2,3,4,5,6,7,8,9,10,11,12,13,14,15);
}

__device__ __forceinline__ v8f wmma_bf16(v16bf a, v16bf b, v8f c) {
  // D = A(16x32 bf16) * B(32x16 bf16) + C(16x16 f32)
  return __builtin_amdgcn_wmma_f32_16x16x32_bf16(false, a, false, b, (short)0, c,
                                                 false, false);
}

// Butterfly max across each 16-lane half (lanes 0-15 / 16-31 independent),
// all-VALU: DPP8 for xor1/2/4, v_permlane16_b32 for xor8 (row-relative).
#define DPP8SEL(a,b,c,d,e,f,g,h) \
  ((a)|((b)<<3)|((c)<<6)|((d)<<9)|((e)<<12)|((f)<<15)|((g)<<18)|((h)<<21))

__device__ __forceinline__ float fmax_bcast16(float v) {
  int t;
  t = __builtin_amdgcn_mov_dpp8(__float_as_int(v), DPP8SEL(1,0,3,2,5,4,7,6));
  v = fmaxf(v, __int_as_float(t));
  t = __builtin_amdgcn_mov_dpp8(__float_as_int(v), DPP8SEL(2,3,0,1,6,7,4,5));
  v = fmaxf(v, __int_as_float(t));
  t = __builtin_amdgcn_mov_dpp8(__float_as_int(v), DPP8SEL(4,5,6,7,0,1,2,3));
  v = fmaxf(v, __int_as_float(t));
  t = __builtin_amdgcn_permlane16(__float_as_int(v), __float_as_int(v),
                                  0xFEDCBA98, 0x76543210, false, false);
  v = fmaxf(v, __int_as_float(t));
  return v;
}

// ---------------------------------------------------------------------------
// TDM: 2D tile load Global -> LDS with row padding (see D# layout, ISA ch.8).
// ---------------------------------------------------------------------------
__device__ __forceinline__ void tdm_load_2d(unsigned int lds_addr, const void* gptr,
                                            unsigned int tensor_d0, unsigned int tensor_d1,
                                            unsigned int tile_d0,   unsigned int tile_d1,
                                            unsigned int stride0,
                                            unsigned int pad_interval, unsigned int pad_amount)
{
  unsigned long long ga = (unsigned long long)gptr;
  u32x4 g0;
  g0[0] = 1u;                                   // count=1 (valid user descriptor)
  g0[1] = lds_addr;                             // D#.lds_addr
  g0[2] = (unsigned int)ga;                     // global_addr[31:0]
  g0[3] = (unsigned int)((ga >> 32) & 0x01ffffffu) | (2u << 30);  // addr[56:32] | type=2
  i32x8 g1;
  g1[0] = (int)((1u << 16) |                    // data_size = 1 (2 bytes)
                (1u << 20) |                    // pad_enable
                (pad_interval << 22) |
                (pad_amount << 25));
  g1[1] = (int)((tensor_d0 & 0xffffu) << 16);   // tensor_dim0[15:0] at bits 63:48
  g1[2] = (int)(((tensor_d0 >> 16) & 0xffffu) | ((tensor_d1 & 0xffffu) << 16));
  g1[3] = (int)(((tensor_d1 >> 16) & 0xffffu) | ((tile_d0 & 0xffffu) << 16));
  g1[4] = (int)(tile_d1 & 0xffffu);             // tile_dim1; tile_dim2 = 0
  g1[5] = (int)stride0;                         // tensor_dim0_stride[31:0]
  g1[6] = 0;
  g1[7] = 0;
  i32x4 gz = {0, 0, 0, 0};
#if __clang_major__ >= 23
  i32x8 gz8 = {0, 0, 0, 0, 0, 0, 0, 0};
  __builtin_amdgcn_tensor_load_to_lds(g0, g1, gz, gz, gz8, 0);
#else
  __builtin_amdgcn_tensor_load_to_lds(g0, g1, gz, gz, 0);
#endif
}

// ---------------------------------------------------------------------------
// Tiled GEMM: Y[M,128] = (X[M,K] @ W[K,128] + bias) * scale
// block = 128 threads (4 waves), 64 rows/block; wave: 16 rows x 128 cols.
// TRANSPOSE=true: write bf16 output as Yt[batch][n][row%LKV] (for V^T).
// ---------------------------------------------------------------------------
template <typename OutT, bool TRANSPOSE>
__global__ __launch_bounds__(128)
void proj_gemm_kernel(const float* __restrict__ X,
                      const float* __restrict__ W,
                      const float* __restrict__ bias,
                      OutT* __restrict__ Y,
                      int M, int K, float scale)
{
  __shared__ bf16 WTc[128][40];   // W chunk transposed: [n][k 0..31], +8 pad

  const int tid   = threadIdx.x;
  const int lane  = tid & 31;
  const int wave  = tid >> 5;
  const int l16   = lane & 15;
  const int khalf = lane >> 4;

  const int row0 = blockIdx.x * 64 + wave * 16;
  const int arow = row0 + l16;          // A-fragment row for this lane

  v8f acc[8];
#pragma unroll
  for (int i = 0; i < 8; ++i)
#pragma unroll
    for (int j = 0; j < 8; ++j) acc[i][j] = 0.0f;

  const int nchunks = K >> 5;
  for (int kc = 0; kc < nchunks; ++kc) {
    const int kbase = kc * 32;

    __syncthreads();   // protect WTc reuse from previous iteration
#pragma unroll
    for (int i = 0; i < 8; ++i) {
      int idx4 = tid + i * 128;            // 0..1023 float4 chunks
      int kk   = idx4 >> 5;                // 0..31
      int nb   = (idx4 & 31) * 4;          // 0..124
      float4 w4 = *(const float4*)(W + (size_t)(kbase + kk) * 128 + nb);
      WTc[nb + 0][kk] = (bf16)w4.x;
      WTc[nb + 1][kk] = (bf16)w4.y;
      WTc[nb + 2][kk] = (bf16)w4.z;
      WTc[nb + 3][kk] = (bf16)w4.w;
    }
    __syncthreads();

    // A fragment (16x32 bf16): lanes 0-15 K{0-7,16-23}, lanes 16-31 K{8-15,24-31}
    const float* xr = X + (size_t)arow * K + kbase + khalf * 8;
    float4 x0 = *(const float4*)(xr);
    float4 x1 = *(const float4*)(xr + 4);
    float4 x2 = *(const float4*)(xr + 16);
    float4 x3 = *(const float4*)(xr + 20);
    v16bf a;
    a[0]=(bf16)x0.x; a[1]=(bf16)x0.y; a[2]=(bf16)x0.z; a[3]=(bf16)x0.w;
    a[4]=(bf16)x1.x; a[5]=(bf16)x1.y; a[6]=(bf16)x1.z; a[7]=(bf16)x1.w;
    a[8]=(bf16)x2.x; a[9]=(bf16)x2.y; a[10]=(bf16)x2.z; a[11]=(bf16)x2.w;
    a[12]=(bf16)x3.x; a[13]=(bf16)x3.y; a[14]=(bf16)x3.z; a[15]=(bf16)x3.w;

#pragma unroll
    for (int nt = 0; nt < 8; ++nt) {
      const bf16* wp = &WTc[nt * 16 + l16][khalf * 8];
      v8bf blo = *(const v8bf*)(wp);
      v8bf bhi = *(const v8bf*)(wp + 16);
      acc[nt] = wmma_bf16(a, cat8(blo, bhi), acc[nt]);
    }
  }

  // Epilogue: C layout -> row = r + 8*khalf, col = l16 (per N-tile)
  if constexpr (TRANSPOSE) {
    const int rr0   = row0 + 8 * khalf;       // 8 consecutive rows from here
    const int batch = rr0 / LKV_;
    const int kv0   = rr0 % LKV_;
#pragma unroll
    for (int nt = 0; nt < 8; ++nt) {
      int n = nt * 16 + l16;
      float bs = bias[n];
      v8bf pk;
#pragma unroll
      for (int r = 0; r < 8; ++r)
        pk[r] = (bf16)((acc[nt][r] + bs) * scale);
      *(v8bf*)((bf16*)Y + (size_t)batch * 128 * LKV_ + (size_t)n * LKV_ + kv0) = pk;
    }
  } else {
#pragma unroll
    for (int nt = 0; nt < 8; ++nt) {
      int n = nt * 16 + l16;
      float bs = bias[n];
#pragma unroll
      for (int r = 0; r < 8; ++r) {
        int rr = row0 + r + 8 * khalf;
        float v = (acc[nt][r] + bs) * scale;
        Y[(size_t)rr * 128 + n] = (OutT)v;
      }
    }
  }
}

// ---------------------------------------------------------------------------
// Flash attention: O = softmax(Q K^T) V   (Q pre-scaled by 1/sqrt(128))
// Qb,Kb bf16 [B*L,128] row-major; Vt bf16 [B][128][LKV] (pre-transposed).
// KV tiles double-buffered into LDS via TDM; softmax denominator accumulated
// with an all-ones WMMA B tile (no cross-lane sum reduction needed).
// ---------------------------------------------------------------------------
__global__ __launch_bounds__(128)
void flash_attn_kernel(const bf16* __restrict__ Qb,
                       const bf16* __restrict__ Kb,
                       const bf16* __restrict__ Vt,
                       float* __restrict__ O)
{
  __shared__ bf16 Ks [2][64][136];   // K tile row-major, +8 pad (272B stride)
  __shared__ bf16 VTs[2][128][72];   // V^T tile [d][kv], +8 pad (144B stride)
  __shared__ bf16 Ps [4][16][72];    // per-wave P scratch (C->A relayout)

  const int tid   = threadIdx.x;
  const int lane  = tid & 31;
  const int wave  = tid >> 5;
  const int l16   = lane & 15;
  const int khalf = lane >> 4;

  const int qrow0_g = blockIdx.x * 64;             // global row in [0, B*LQ)
  const int batch   = qrow0_g / LQ_;
  const int wrow0   = qrow0_g + wave * 16;

  const bf16* Kbase  = Kb + (size_t)batch * LKV_ * 128;
  const bf16* Vtbase = Vt + (size_t)batch * 128 * LKV_;

  // Preload Q A-fragments (4 chunks over D=128), held in registers
  v16bf aq[4];
  {
    const bf16* qr = Qb + (size_t)(wrow0 + l16) * 128;
#pragma unroll
    for (int kc = 0; kc < 4; ++kc) {
      const bf16* p = qr + kc * 32 + khalf * 8;
      aq[kc] = cat8(*(const v8bf*)(p), *(const v8bf*)(p + 16));
    }
  }

  v16bf ones;
#pragma unroll
  for (int e = 0; e < 16; ++e) ones[e] = (bf16)1.0f;

  v8f o[8], o9;                      // o9: softmax denominator accumulator
#pragma unroll
  for (int j = 0; j < 8; ++j) o9[j] = 0.0f;
#pragma unroll
  for (int i = 0; i < 8; ++i)
#pragma unroll
    for (int j = 0; j < 8; ++j) o[i][j] = 0.0f;

  float mrow[8];
#pragma unroll
  for (int r = 0; r < 8; ++r) mrow[r] = -__builtin_inff();

  const int NT = LKV_ / 64;

  // Prologue: DMA tile 0 into buffer 0 (wave 0 issues; EXEC ignored by TDM)
  if (wave == 0) {
    tdm_load_2d((unsigned int)(size_t)&Ks[0][0][0],  Kbase,
                128u, (unsigned)LKV_, 128u, 64u, 128u, 5u, 3u);
    tdm_load_2d((unsigned int)(size_t)&VTs[0][0][0], Vtbase,
                (unsigned)LKV_, 128u, 64u, 128u, (unsigned)LKV_, 4u, 3u);
  }

  for (int it = 0; it < NT; ++it) {
    const int buf = it & 1;
    if (wave == 0) {
      if (it + 1 < NT) {
        const int kvn = (it + 1) * 64;
        tdm_load_2d((unsigned int)(size_t)&Ks[buf ^ 1][0][0],
                    Kbase + (size_t)kvn * 128,
                    128u, (unsigned)LKV_, 128u, 64u, 128u, 5u, 3u);
        tdm_load_2d((unsigned int)(size_t)&VTs[buf ^ 1][0][0],
                    Vtbase + kvn,
                    (unsigned)LKV_, 128u, 64u, 128u, (unsigned)LKV_, 4u, 3u);
        __builtin_amdgcn_s_wait_tensorcnt(2);   // current tile's pair done
      } else {
        __builtin_amdgcn_s_wait_tensorcnt(0);
      }
    }
    __syncthreads();

    const bf16 (*KsT)[136] = Ks[buf];
    const bf16 (*VTsT)[72] = VTs[buf];

    // --- S = Q K^T : 4 tiles of 16x16, f32 accumulate ---
    v8f s[4];
#pragma unroll
    for (int i = 0; i < 4; ++i)
#pragma unroll
      for (int j = 0; j < 8; ++j) s[i][j] = 0.0f;
#pragma unroll
    for (int kc = 0; kc < 4; ++kc) {
#pragma unroll
      for (int nt = 0; nt < 4; ++nt) {
        const bf16* kp = &KsT[nt * 16 + l16][kc * 32 + khalf * 8];
        s[nt] = wmma_bf16(aq[kc],
                          cat8(*(const v8bf*)kp, *(const v8bf*)(kp + 16)),
                          s[nt]);
      }
    }

    // --- online softmax: row max (all-VALU butterfly), P = exp(S - m) ---
    float alpha[8];
#pragma unroll
    for (int r = 0; r < 8; ++r) {
      float mx = fmaxf(fmaxf(s[0][r], s[1][r]), fmaxf(s[2][r], s[3][r]));
      mx = fmax_bcast16(mx);
      float mnew = fmaxf(mrow[r], mx);
      alpha[r] = __expf(mrow[r] - mnew);
      mrow[r]  = mnew;
    }
#pragma unroll
    for (int nt = 0; nt < 4; ++nt)
#pragma unroll
      for (int r = 0; r < 8; ++r)
        s[nt][r] = __expf(s[nt][r] - mrow[r]);
#pragma unroll
    for (int nt = 0; nt < 8; ++nt)
#pragma unroll
      for (int r = 0; r < 8; ++r) o[nt][r] *= alpha[r];
#pragma unroll
    for (int r = 0; r < 8; ++r) o9[r] *= alpha[r];

    // --- relayout P: C layout -> bf16 A fragments via per-wave LDS ---
#pragma unroll
    for (int nt = 0; nt < 4; ++nt)
#pragma unroll
      for (int r = 0; r < 8; ++r)
        Ps[wave][r + 8 * khalf][nt * 16 + l16] = (bf16)s[nt][r];
    asm volatile("s_wait_dscnt 0" ::: "memory");
    __builtin_amdgcn_wave_barrier();

    // --- O += P V ; denominator += P * ones ---
#pragma unroll
    for (int kc2 = 0; kc2 < 2; ++kc2) {
      const bf16* pp = &Ps[wave][l16][kc2 * 32 + khalf * 8];
      v16bf ap = cat8(*(const v8bf*)pp, *(const v8bf*)(pp + 16));
#pragma unroll
      for (int nt = 0; nt < 8; ++nt) {
        const bf16* vp = &VTsT[nt * 16 + l16][kc2 * 32 + khalf * 8];
        o[nt] = wmma_bf16(ap,
                          cat8(*(const v8bf*)vp, *(const v8bf*)(vp + 16)),
                          o[nt]);
      }
      o9 = wmma_bf16(ap, ones, o9);
    }
    __syncthreads();   // all waves done reading this buffer
  }

  // --- epilogue: normalize by denominator and store f32 attn ---
  float linv[8];
#pragma unroll
  for (int r = 0; r < 8; ++r) linv[r] = 1.0f / o9[r];
#pragma unroll
  for (int nt = 0; nt < 8; ++nt) {
    int n = nt * 16 + l16;
#pragma unroll
    for (int r = 0; r < 8; ++r) {
      int rr = wrow0 + r + 8 * khalf;
      O[(size_t)rr * 128 + n] = o[nt][r] * linv[r];
    }
  }
}

// ---------------------------------------------------------------------------
extern "C" void kernel_launch(void* const* d_in, const int* in_sizes, int n_in,
                              void* d_out, int out_size, void* d_ws, size_t ws_size,
                              hipStream_t stream) {
  (void)in_sizes; (void)n_in; (void)out_size; (void)ws_size;

  const float* ligand = (const float*)d_in[0];
  const float* pocket = (const float*)d_in[1];
  const float* Wq = (const float*)d_in[2];
  const float* bq = (const float*)d_in[3];
  const float* Wk = (const float*)d_in[4];
  const float* bk = (const float*)d_in[5];
  const float* Wv = (const float*)d_in[6];
  const float* bv = (const float*)d_in[7];
  const float* Wo = (const float*)d_in[8];
  const float* bo = (const float*)d_in[9];
  float* out = (float*)d_out;

  const int M = B_ * LQ_;                        // 16384 rows
  char* ws = (char*)d_ws;
  bf16*  Qb   = (bf16*)ws;                       // 4 MB
  bf16*  Kb   = Qb + (size_t)M * 128;            // 4 MB
  bf16*  Vtb  = Kb + (size_t)M * 128;            // 4 MB  (transposed: [B][128][LKV])
  float* attn = (float*)(Vtb + (size_t)M * 128); // 8 MB

  const float qscale = 0.08838834764831845f;     // 1/sqrt(128)
  dim3 blk(128);
  dim3 grd(M / 64);

  proj_gemm_kernel<bf16, false><<<grd, blk, 0, stream>>>(ligand, Wq, bq, Qb,  M, DIN_, qscale);
  proj_gemm_kernel<bf16, false><<<grd, blk, 0, stream>>>(pocket, Wk, bk, Kb,  M, DIN_, 1.0f);
  proj_gemm_kernel<bf16, true ><<<grd, blk, 0, stream>>>(pocket, Wv, bv, Vtb, M, DIN_, 1.0f);
  flash_attn_kernel<<<grd, blk, 0, stream>>>(Qb, Kb, Vtb, attn);
  proj_gemm_kernel<float, false><<<grd, blk, 0, stream>>>(attn, Wo, bo, out, M, DOUT_, 1.0f);
}
